// DeepMonocularModel_40046275068461
// MI455X (gfx1250) — matrix operand order repaired and checked by reading, not compiled, not run on hardware
//
#include <hip/hip_runtime.h>
#include <cstddef>

#define BB 256
#define DD 768
#define NN 256
#define LL 2
#define HH 8
#define TT 20
#define HD 96
#define D4 3072
#define MTOK (BB * NN)   // 65536

typedef __attribute__((ext_vector_type(16))) __bf16 bf16x16;
typedef __attribute__((ext_vector_type(8)))  float  f32x8;
typedef int v4i __attribute__((vector_size(16)));   // matches builtin param type

union FragAB { bf16x16 v; uint4 u[2]; };
union U4H8   { uint4 u; unsigned short h[8]; };

#define AS_GLOBAL __attribute__((address_space(1)))
#define AS_LDS    __attribute__((address_space(3)))

#if defined(__gfx1250__) && __has_builtin(__builtin_amdgcn_global_load_async_to_lds_b128) && \
    __has_builtin(__builtin_amdgcn_s_wait_asynccnt)
#define HAVE_ASYNC_LDS 1
#else
#define HAVE_ASYNC_LDS 0
#endif

__device__ __forceinline__ float bf2f(unsigned short u) {
    unsigned int x = ((unsigned int)u) << 16;
    return __builtin_bit_cast(float, x);
}
__device__ __forceinline__ unsigned short f2bf(float f) {
    unsigned int x = __builtin_bit_cast(unsigned int, f);
    x += 0x7FFFu + ((x >> 16) & 1u);           // round-to-nearest-even
    return (unsigned short)(x >> 16);
}
__device__ __forceinline__ float gelu_exact(float x) {
    return 0.5f * x * (1.0f + erff(x * 0.70710678118654752f));
}
// blockDim.x == 256 reductions
__device__ __forceinline__ float block_sum_256(float v, float* red) {
    int t = threadIdx.x;
    red[t] = v; __syncthreads();
    for (int off = 128; off > 0; off >>= 1) {
        if (t < off) red[t] += red[t + off];
        __syncthreads();
    }
    float r = red[0]; __syncthreads();
    return r;
}

// ---------------------------------------------------------------------------
// WMMA bf16 GEMM:  C(M x N) = epilogue(A(M x K) @ W(K x N) + bias)
// Double-buffered LDS software pipeline. A tile staged via gfx1250 async
// global->LDS copies (ASYNCcnt) when available, else register pipeline.
// EPI: 0 = store f32, 1 = store bf16, 2 = Cf += (residual), 3 = gelu->bf16, 4 = gelu->f32
// Requires: M % BM == 0, N % BN == 0, K % 32 == 0 (true for all call sites).
// ---------------------------------------------------------------------------
template<int BM, int BN, int WM, int WN, int EPI>
__global__ __launch_bounds__(256) void gemm_wmma(
    const unsigned short* __restrict__ A, int lda,
    const unsigned short* __restrict__ W, int ldb,
    const float* __restrict__ bias,
    float* __restrict__ Cf, unsigned short* __restrict__ Cb, int ldc,
    int K)
{
    constexpr int KT = 32;
    constexpr int KP = KT + 8;                  // padded row (halfs) -> 80B stride
    constexpr int WAVES_N = BN / WN;
    constexpr int TM = WM / 16;
    constexpr int TN = WN / 16;
    constexpr int APT = (BM * (KT / 8)) / 256;  // A 16B-chunks per thread
    constexpr int BPT = (KT * (BN / 8)) / 256;  // W 16B-chunks per thread

    __shared__ __align__(16) unsigned short As[2][BM * KP];
    __shared__ __align__(16) unsigned short Bs[2][BN * KP];

    const int tid  = threadIdx.x;
    const int lane = tid & 31;
    const int wid  = tid >> 5;
    const int g    = lane >> 4;                 // half-wave group
    const int lr   = lane & 15;
    const int wm   = (wid / WAVES_N) * WM;
    const int wn   = (wid % WAVES_N) * WN;
    const int m0   = blockIdx.y * BM;
    const int n0   = blockIdx.x * BN;

    // staging coordinates (compile-time trip counts, fully unrolled)
    int arow[APT], acol[APT];
#pragma unroll
    for (int p = 0; p < APT; p++) {
        int c = tid + p * 256; arow[p] = c >> 2; acol[p] = (c & 3) * 8;
    }
    int bkk[BPT], bnn[BPT];
#pragma unroll
    for (int p = 0; p < BPT; p++) {
        int c = tid + p * 256; bkk[p] = c / (BN / 8); bnn[p] = (c % (BN / 8)) * 8;
    }

    f32x8 acc[TM][TN];
#pragma unroll
    for (int i = 0; i < TM; i++)
#pragma unroll
        for (int j = 0; j < TN; j++)
#pragma unroll
            for (int r = 0; r < 8; r++) acc[i][j][r] = 0.0f;

    uint4 rb[BPT];
    auto loadB = [&](int k0) {
#pragma unroll
        for (int p = 0; p < BPT; p++)
            rb[p] = *(const uint4*)(W + (size_t)(k0 + bkk[p]) * ldb + n0 + bnn[p]);
    };
    auto storeB = [&](int buf) {                // transpose scatter -> Bs[n][k]
#pragma unroll
        for (int p = 0; p < BPT; p++) {
            U4H8 vv; vv.u = rb[p];
#pragma unroll
            for (int e = 0; e < 8; e++)
                Bs[buf][(bnn[p] + e) * KP + bkk[p]] = vv.h[e];
        }
    };
    auto compute = [&](int buf) {
        FragAB af[TM], bfr[TN];
#pragma unroll
        for (int i = 0; i < TM; i++) {
            const unsigned short* p = &As[buf][(wm + i * 16 + lr) * KP];
            af[i].u[0] = *(const uint4*)(p + 8 * g);        // K = 8g..8g+7
            af[i].u[1] = *(const uint4*)(p + 16 + 8 * g);   // K = 16+8g..+7
        }
#pragma unroll
        for (int j = 0; j < TN; j++) {
            const unsigned short* p = &Bs[buf][(wn + j * 16 + lr) * KP];
            bfr[j].u[0] = *(const uint4*)(p + 8 * g);       // K = 8g..8g+7
            bfr[j].u[1] = *(const uint4*)(p + 16 + 8 * g);  // K = 16+8g..+7
        }
#pragma unroll
        for (int i = 0; i < TM; i++)
#pragma unroll
            for (int j = 0; j < TN; j++)
                acc[i][j] = __builtin_amdgcn_wmma_f32_16x16x32_bf16(
                    false, af[i].v, false, bfr[j].v, (short)0, acc[i][j], false, false);
    };
    auto prefetch_ahead = [&](int it, int nsteps) {
        if (it + 4 < nsteps) {
            int kp = (it + 4) * KT;
            __builtin_prefetch((const void*)(A + (size_t)(m0 + arow[0]) * lda + kp + acol[0]), 0, 1);
            __builtin_prefetch((const void*)(W + (size_t)(kp + bkk[0]) * ldb + n0 + bnn[0]), 0, 1);
        }
    };

    const int nsteps = K / KT;

#if HAVE_ASYNC_LDS
    auto asyncA = [&](int k0, int buf) {        // gfx1250 async global->LDS (ASYNCcnt)
#pragma unroll
        for (int p = 0; p < APT; p++)
            __builtin_amdgcn_global_load_async_to_lds_b128(
                (AS_GLOBAL v4i*)(v4i*)(A + (size_t)(m0 + arow[p]) * lda + k0 + acol[p]),
                (AS_LDS v4i*)(v4i*)(&As[buf][arow[p] * KP + acol[p]]),
                0, 0);
    };
    asyncA(0, 0);
    loadB(0);
    for (int it = 0; it < nsteps; ++it) {
        const int buf = it & 1;
        storeB(buf);
        if (it + 1 < nsteps) loadB((it + 1) * KT);
        prefetch_ahead(it, nsteps);
        __builtin_amdgcn_s_wait_asynccnt(0);    // A(it) landed in LDS
        __syncthreads();                        // all waves' tiles visible
        if (it + 1 < nsteps) asyncA((it + 1) * KT, 1 - buf);  // safe: buffer's last
                                                              // readers finished before barrier
        compute(buf);
    }
#else
    uint4 ra[APT];
    auto loadA = [&](int k0) {
#pragma unroll
        for (int p = 0; p < APT; p++)
            ra[p] = *(const uint4*)(A + (size_t)(m0 + arow[p]) * lda + k0 + acol[p]);
    };
    auto storeA = [&](int buf) {
#pragma unroll
        for (int p = 0; p < APT; p++)
            *(uint4*)(&As[buf][arow[p] * KP + acol[p]]) = ra[p];
    };
    loadA(0); loadB(0);
    for (int it = 0; it < nsteps; ++it) {
        const int buf = it & 1;
        storeA(buf); storeB(buf);
        if (it + 1 < nsteps) { loadA((it + 1) * KT); loadB((it + 1) * KT); }
        prefetch_ahead(it, nsteps);
        __syncthreads();
        compute(buf);
    }
#endif

    // ---- epilogue (C layout: elem r -> M = r + 8g, N = lr) ----
#pragma unroll
    for (int j = 0; j < TN; j++) {
        int n = n0 + wn + j * 16 + lr;
        float bv = bias[n];
#pragma unroll
        for (int i = 0; i < TM; i++) {
#pragma unroll
            for (int r = 0; r < 8; r++) {
                int m = m0 + wm + i * 16 + 8 * g + r;
                float v = acc[i][j][r] + bv;
                size_t off = (size_t)m * ldc + n;
                if (EPI == 0)      Cf[off] = v;
                else if (EPI == 1) Cb[off] = f2bf(v);
                else if (EPI == 2) Cf[off] += v;
                else if (EPI == 3) Cb[off] = f2bf(gelu_exact(v));
                else               Cf[off] = gelu_exact(v);
            }
        }
    }
}

// ---------------------------------------------------------------------------
// tokens = transpose(feats,(0,2,1)) + pos_enc  -> bf16, LDS tiled transpose
// grid (B, D/64, N/64), block 256
// ---------------------------------------------------------------------------
__global__ __launch_bounds__(256) void tok_prep_kernel(
    const float* __restrict__ feats, const float* __restrict__ pos,
    unsigned short* __restrict__ tok)
{
    __shared__ float tile[64][65];
    int b = blockIdx.x, d0 = blockIdx.y * 64, n0 = blockIdx.z * 64;
    int t = threadIdx.x;
    int tn = t & 63, ti = t >> 6;            // 4 rows/pass, coalesced along n
    for (int r = 0; r < 64; r += 4)
        tile[ti + r][tn] = feats[((size_t)b * DD + d0 + ti + r) * NN + n0 + tn];
    __syncthreads();
    int td = t & 63, tj = t >> 6;            // coalesced along d
    for (int r = 0; r < 64; r += 4) {
        int n = n0 + tj + r, d = d0 + td;
        tok[((size_t)b * NN + n) * DD + d] =
            f2bf(tile[td][tj + r] + pos[(size_t)n * DD + d]);
    }
}

__global__ void f2bf_kernel(const float* __restrict__ s, unsigned short* __restrict__ d, int n) {
    int i = blockIdx.x * 256 + threadIdx.x;
    if (i < n) d[i] = f2bf(s[i]);
}

// ---------------------------------------------------------------------------
// context = concat(onehot, past_flat, 0) @ w_qinit + b_qinit ; cp init
// grid B, block 256
// ---------------------------------------------------------------------------
__global__ __launch_bounds__(256) void context_init_kernel(
    const float* __restrict__ past, const float* __restrict__ w_qinit,
    const float* __restrict__ b_qinit, const int* __restrict__ intent,
    float* __restrict__ x, float* __restrict__ cp)
{
    int b = blockIdx.x, t = threadIdx.x;
    int cls = intent[b] - 1; cls = cls < 0 ? 0 : (cls > 2 ? 2 : cls);
#pragma unroll
    for (int e = 0; e < 3; e++) {
        int d = t + 256 * e;
        float acc = b_qinit[d] + w_qinit[(size_t)cls * DD + d];
        for (int j = 0; j < 96; j++)
            acc += past[b * 96 + j] * w_qinit[(size_t)(3 + j) * DD + d];
        x[(size_t)b * DD + d] = acc;
    }
    if (t < 2) cp[b * 2 + t] = past[b * 96 + 90 + t];   // past[:, -1, 0:2]
}

// ---------------------------------------------------------------------------
// x += relu(LN(cp @ w_pp + b_pp)) + te   (grid B, block 256)
// ---------------------------------------------------------------------------
__global__ __launch_bounds__(256) void step_init_kernel(
    float* __restrict__ x, const float* __restrict__ cp,
    const float* __restrict__ w_pp, const float* __restrict__ b_pp,
    const float* __restrict__ g_pp, const float* __restrict__ be_pp,
    const float* __restrict__ te)
{
    __shared__ float red[256];
    int b = blockIdx.x, t = threadIdx.x;
    float c0 = cp[b * 2], c1 = cp[b * 2 + 1];
    float v[3]; float s = 0.0f;
#pragma unroll
    for (int e = 0; e < 3; e++) {
        int d = t + 256 * e;
        v[e] = c0 * w_pp[d] + c1 * w_pp[DD + d] + b_pp[d];
        s += v[e];
    }
    float mean = block_sum_256(s, red) * (1.0f / DD);
    float q = 0.0f;
#pragma unroll
    for (int e = 0; e < 3; e++) { float dv = v[e] - mean; q += dv * dv; }
    float var = block_sum_256(q, red) * (1.0f / DD);
    float inv = rsqrtf(var + 1e-5f);
#pragma unroll
    for (int e = 0; e < 3; e++) {
        int d = t + 256 * e;
        float pq = (v[e] - mean) * inv * g_pp[d] + be_pp[d];
        pq = fmaxf(pq, 0.0f);
        x[(size_t)b * DD + d] += pq + te[d];
    }
}

// ---------------------------------------------------------------------------
// out_bf16 = LN(x) * g + b   (grid B, block 256)
// ---------------------------------------------------------------------------
__global__ __launch_bounds__(256) void ln_bf16_kernel(
    const float* __restrict__ x, const float* __restrict__ gam,
    const float* __restrict__ bet, unsigned short* __restrict__ out)
{
    __shared__ float red[256];
    int b = blockIdx.x, t = threadIdx.x;
    float v[3]; float s = 0.0f;
#pragma unroll
    for (int e = 0; e < 3; e++) { v[e] = x[(size_t)b * DD + t + 256 * e]; s += v[e]; }
    float mean = block_sum_256(s, red) * (1.0f / DD);
    float q = 0.0f;
#pragma unroll
    for (int e = 0; e < 3; e++) { float dv = v[e] - mean; q += dv * dv; }
    float var = block_sum_256(q, red) * (1.0f / DD);
    float inv = rsqrtf(var + 1e-5f);
#pragma unroll
    for (int e = 0; e < 3; e++) {
        int d = t + 256 * e;
        out[(size_t)b * DD + d] = f2bf((v[e] - mean) * inv * gam[d] + bet[d]);
    }
}

// ---------------------------------------------------------------------------
// Single-query attention per (b,h): s = qK^T/sqrt(HD), softmax, o = aV -> bf16
// grid B*H, block 128.  K dot: per-thread row stream; V accumulate: coalesced.
// ---------------------------------------------------------------------------
__global__ __launch_bounds__(128) void attn_kernel(
    const float* __restrict__ q, const unsigned short* __restrict__ Kc,
    const unsigned short* __restrict__ Vc, unsigned short* __restrict__ o16)
{
    __shared__ float qs[HD];
    __shared__ float a_s[NN];
    __shared__ float red[128];
    int bid = blockIdx.x, b = bid >> 3, h = bid & 7;
    int t = threadIdx.x;
    if (t < HD) qs[t] = q[(size_t)b * DD + h * HD + t];
    __syncthreads();

    float sc[2];
#pragma unroll
    for (int r = 0; r < 2; r++) {
        int n = t + 128 * r;
        const unsigned short* kr = Kc + ((size_t)(b * NN + n)) * DD + h * HD;
        float acc = 0.0f;
#pragma unroll
        for (int c = 0; c < HD / 8; c++) {
            U4H8 vv; vv.u = *(const uint4*)(kr + c * 8);
#pragma unroll
            for (int e = 0; e < 8; e++) acc += qs[c * 8 + e] * bf2f(vv.h[e]);
        }
        sc[r] = acc * 0.10206207261596577f;   // 1/sqrt(96)
    }
    // block max over 256 scores
    red[t] = fmaxf(sc[0], sc[1]); __syncthreads();
    for (int off = 64; off > 0; off >>= 1) {
        if (t < off) red[t] = fmaxf(red[t], red[t + off]);
        __syncthreads();
    }
    float mx = red[0]; __syncthreads();
    float e0 = __expf(sc[0] - mx), e1 = __expf(sc[1] - mx);
    a_s[t] = e0; a_s[t + 128] = e1;
    red[t] = e0 + e1; __syncthreads();
    for (int off = 64; off > 0; off >>= 1) {
        if (t < off) red[t] += red[t + off];
        __syncthreads();
    }
    float inv = 1.0f / red[0];
    if (t < HD) {
        float acc = 0.0f;
        for (int n = 0; n < NN; n++)
            acc += a_s[n] * bf2f(Vc[((size_t)(b * NN + n)) * DD + h * HD + t]);
        o16[(size_t)b * DD + h * HD + t] = f2bf(acc * inv);
    }
}

// ---------------------------------------------------------------------------
// delta = g @ w_dec2 + b_dec2 ; cp += delta ; out[b,t,:] = cp  (grid B, block 256)
// ---------------------------------------------------------------------------
__global__ __launch_bounds__(256) void dec2_kernel(
    const float* __restrict__ gbuf, const float* __restrict__ w_dec2,
    const float* __restrict__ b_dec2, float* __restrict__ cp,
    float* __restrict__ out, int tstep)
{
    __shared__ float red[256];
    int b = blockIdx.x, t = threadIdx.x;
    float p0 = 0.0f, p1 = 0.0f;
#pragma unroll
    for (int e = 0; e < 3; e++) {
        int d = t + 256 * e;
        float gv = gbuf[(size_t)b * DD + d];
        p0 += gv * w_dec2[d * 2 + 0];
        p1 += gv * w_dec2[d * 2 + 1];
    }
    p0 = block_sum_256(p0, red);
    p1 = block_sum_256(p1, red);
    if (t == 0) {
        float n0 = cp[b * 2 + 0] + p0 + b_dec2[0];
        float n1 = cp[b * 2 + 1] + p1 + b_dec2[1];
        cp[b * 2 + 0] = n0; cp[b * 2 + 1] = n1;
        out[((size_t)b * TT + tstep) * 2 + 0] = n0;
        out[((size_t)b * TT + tstep) * 2 + 1] = n1;
    }
}

// ---------------------------------------------------------------------------
extern "C" void kernel_launch(void* const* d_in, const int* in_sizes, int n_in,
                              void* d_out, int out_size, void* d_ws, size_t ws_size,
                              hipStream_t stream)
{
    const float* feats    = (const float*)d_in[0];
    const float* past     = (const float*)d_in[1];
    const float* w_qinit  = (const float*)d_in[2];
    const float* b_qinit  = (const float*)d_in[3];
    const float* pos_enc  = (const float*)d_in[4];
    const float* w_pp     = (const float*)d_in[5];
    const float* b_pp     = (const float*)d_in[6];
    const float* g_pp     = (const float*)d_in[7];
    const float* be_pp    = (const float*)d_in[8];
    const float* time_emb = (const float*)d_in[9];
    const float* Wq = (const float*)d_in[10]; const float* bq = (const float*)d_in[11];
    const float* Wk = (const float*)d_in[12]; const float* bk = (const float*)d_in[13];
    const float* Wv = (const float*)d_in[14]; const float* bv = (const float*)d_in[15];
    const float* Wo = (const float*)d_in[16]; const float* bo = (const float*)d_in[17];
    const float* ln1_g = (const float*)d_in[18]; const float* ln1_b = (const float*)d_in[19];
    const float* W1 = (const float*)d_in[20]; const float* b1 = (const float*)d_in[21];
    const float* W2 = (const float*)d_in[22]; const float* b2 = (const float*)d_in[23];
    const float* ln2_g = (const float*)d_in[24]; const float* ln2_b = (const float*)d_in[25];
    const float* w_dec1 = (const float*)d_in[26]; const float* b_dec1 = (const float*)d_in[27];
    const float* w_dec2 = (const float*)d_in[28]; const float* b_dec2 = (const float*)d_in[29];
    const int*   intent = (const int*)d_in[30];
    float* out = (float*)d_out;

    char* ws = (char*)d_ws;
    size_t off = 0;
    auto alloc = [&](size_t bytes) -> char* {
        char* p = ws + off;
        off += (bytes + 255) & ~(size_t)255;
        return p;
    };
    unsigned short* tok16   = (unsigned short*)alloc((size_t)MTOK * DD * 2);
    unsigned short* kh16    = (unsigned short*)alloc((size_t)LL * MTOK * DD * 2);
    unsigned short* vh16    = (unsigned short*)alloc((size_t)LL * MTOK * DD * 2);
    unsigned short* wq16    = (unsigned short*)alloc((size_t)LL * DD * DD * 2);
    unsigned short* wk16    = (unsigned short*)alloc((size_t)LL * DD * DD * 2);
    unsigned short* wv16    = (unsigned short*)alloc((size_t)LL * DD * DD * 2);
    unsigned short* wo16    = (unsigned short*)alloc((size_t)LL * DD * DD * 2);
    unsigned short* w1_16   = (unsigned short*)alloc((size_t)LL * DD * D4 * 2);
    unsigned short* w2_16   = (unsigned short*)alloc((size_t)LL * D4 * DD * 2);
    unsigned short* wdec116 = (unsigned short*)alloc((size_t)DD * DD * 2);
    float*          xbuf    = (float*)alloc((size_t)BB * DD * 4);
    unsigned short* xn16    = (unsigned short*)alloc((size_t)BB * DD * 2);
    float*          qbuf    = (float*)alloc((size_t)BB * DD * 4);
    unsigned short* o16     = (unsigned short*)alloc((size_t)BB * DD * 2);
    unsigned short* h1_16   = (unsigned short*)alloc((size_t)BB * D4 * 2);
    float*          gbuf    = (float*)alloc((size_t)BB * DD * 4);
    unsigned short* ctx16   = (unsigned short*)alloc((size_t)BB * DD * 2);
    float*          cp      = (float*)alloc((size_t)BB * 2 * 4);

    auto cvt = [&](const float* s, unsigned short* d, int n) {
        f2bf_kernel<<<dim3((n + 255) / 256), 256, 0, stream>>>(s, d, n);
    };
    cvt(Wq, wq16, LL * DD * DD);
    cvt(Wk, wk16, LL * DD * DD);
    cvt(Wv, wv16, LL * DD * DD);
    cvt(Wo, wo16, LL * DD * DD);
    cvt(W1, w1_16, LL * DD * D4);
    cvt(W2, w2_16, LL * D4 * DD);
    cvt(w_dec1, wdec116, DD * DD);

    tok_prep_kernel<<<dim3(BB, DD / 64, NN / 64), 256, 0, stream>>>(feats, pos_enc, tok16);

    // K/V cache precompute: 65536x768x768 bf16 WMMA GEMMs, store bf16
    for (int l = 0; l < LL; l++) {
        gemm_wmma<128, 128, 64, 32, 1><<<dim3(DD / 128, MTOK / 128), 256, 0, stream>>>(
            tok16, DD, wk16 + (size_t)l * DD * DD, DD, bk + l * DD,
            nullptr, kh16 + (size_t)l * MTOK * DD, DD, DD);
        gemm_wmma<128, 128, 64, 32, 1><<<dim3(DD / 128, MTOK / 128), 256, 0, stream>>>(
            tok16, DD, wv16 + (size_t)l * DD * DD, DD, bv + l * DD,
            nullptr, vh16 + (size_t)l * MTOK * DD, DD, DD);
    }

    context_init_kernel<<<dim3(BB), 256, 0, stream>>>(past, w_qinit, b_qinit, intent, xbuf, cp);

    for (int t = 0; t < TT; t++) {
        step_init_kernel<<<dim3(BB), 256, 0, stream>>>(
            xbuf, cp, w_pp, b_pp, g_pp, be_pp, time_emb + (size_t)t * DD);

        for (int l = 0; l < LL; l++) {
            ln_bf16_kernel<<<dim3(BB), 256, 0, stream>>>(xbuf, ln1_g + l * DD, ln1_b + l * DD, xn16);
            gemm_wmma<64, 128, 32, 32, 0><<<dim3(DD / 128, BB / 64), 256, 0, stream>>>(
                xn16, DD, wq16 + (size_t)l * DD * DD, DD, bq + l * DD,
                qbuf, nullptr, DD, DD);
            attn_kernel<<<dim3(BB * HH), 128, 0, stream>>>(
                qbuf, kh16 + (size_t)l * MTOK * DD, vh16 + (size_t)l * MTOK * DD, o16);
            gemm_wmma<64, 128, 32, 32, 2><<<dim3(DD / 128, BB / 64), 256, 0, stream>>>(
                o16, DD, wo16 + (size_t)l * DD * DD, DD, bo + l * DD,
                xbuf, nullptr, DD, DD);
            ln_bf16_kernel<<<dim3(BB), 256, 0, stream>>>(xbuf, ln2_g + l * DD, ln2_b + l * DD, xn16);
            gemm_wmma<64, 128, 32, 32, 3><<<dim3(D4 / 128, BB / 64), 256, 0, stream>>>(
                xn16, DD, w1_16 + (size_t)l * DD * D4, D4, b1 + l * D4,
                nullptr, h1_16, D4, DD);
            gemm_wmma<64, 128, 32, 32, 2><<<dim3(DD / 128, BB / 64), 256, 0, stream>>>(
                h1_16, D4, w2_16 + (size_t)l * D4 * DD, DD, b2 + l * DD,
                xbuf, nullptr, DD, D4);
        }

        f2bf_kernel<<<dim3((BB * DD + 255) / 256), 256, 0, stream>>>(xbuf, ctx16, BB * DD);
        gemm_wmma<64, 128, 32, 32, 4><<<dim3(DD / 128, BB / 64), 256, 0, stream>>>(
            ctx16, DD, wdec116, DD, b_dec1, gbuf, nullptr, DD, DD);
        dec2_kernel<<<dim3(BB), 256, 0, stream>>>(gbuf, w_dec2, b_dec2, cp, out, t);
    }
    (void)in_sizes; (void)n_in; (void)out_size; (void)ws_size;
}
